// CausalSelfAttention_22488448762374
// MI455X (gfx1250) — matrix operand order repaired and checked by reading, not compiled, not run on hardware
//
#include <hip/hip_runtime.h>

// ---------- types ----------
typedef __bf16 bf16;
typedef __attribute__((ext_vector_type(16))) __bf16 v16bf;
typedef __attribute__((ext_vector_type(8)))  __bf16 v8bf;
typedef __attribute__((ext_vector_type(8)))  float  v8f;
typedef __attribute__((ext_vector_type(4)))  unsigned int u32x4;
typedef __attribute__((ext_vector_type(8)))  int i32x8;
typedef __attribute__((ext_vector_type(4)))  int i32x4;

union BF16x16 { v16bf v; v8bf h[2]; };

__device__ __forceinline__ bf16 f2bf(float x) { return (bf16)x; }

__device__ __forceinline__ v8f wmma_bf16(v16bf a, v16bf b, v8f c) {
    // D = A(16x32 bf16) * B(32x16 bf16) + C(16x16 f32)
    return __builtin_amdgcn_wmma_f32_16x16x32_bf16(
        false, a, false, b, (short)0, c, false, false);
}

__device__ __forceinline__ void store_out(float* p, float v) { *p = v; }
__device__ __forceinline__ void store_out(bf16*  p, float v) { *p = f2bf(v); }

// ---------- constants ----------
#define BB   4
#define TT   2048
#define CC   1024
#define HH   16
#define HD   64
#define MM   (BB * TT)       // 8192

// ======================================================================
// TDM 2-D tile load: global (bf16, row stride = stride0 elements)
//                 -> LDS packed [tile_d1][tile_d0] bf16.
// D# per CDNA5 ISA ch.8: group0 = {count|flags, lds_addr, gaddr_lo,
// gaddr_hi|type=2}; group1 carries data_size=2B, tensor/tile dims,
// dim0 stride; groups 2/3 zero (2-D tile, tile_dim2 = 0 = unused).
// 6-arg builtin form (clang-23 / therock-10.0 headers).
// ======================================================================
__device__ __forceinline__ void tdm_load_2d(unsigned lds_off, const bf16* gptr,
                                            unsigned tensor_d0, unsigned tensor_d1,
                                            unsigned tile_d0, unsigned tile_d1,
                                            unsigned stride0) {
    unsigned long long ga = (unsigned long long)(uintptr_t)gptr;
    u32x4 g0;
    g0[0] = 1u;                                   // count=1 (valid user D#)
    g0[1] = lds_off;                              // lds_addr (bytes)
    g0[2] = (unsigned)ga;                         // global_addr[31:0]
    g0[3] = (unsigned)(ga >> 32) | (2u << 30);    // global_addr[56:32] | type=2
    i32x8 g1;
    g1[0] = (int)(1u << 16);                                          // data_size=1 -> 2 bytes
    g1[1] = (int)((tensor_d0 & 0xFFFFu) << 16);                       // tensor_dim0[15:0]
    g1[2] = (int)((tensor_d0 >> 16) | ((tensor_d1 & 0xFFFFu) << 16)); // dim0 hi | dim1 lo
    g1[3] = (int)((tensor_d1 >> 16) | (tile_d0 << 16));               // dim1 hi | tile_dim0
    g1[4] = (int)tile_d1;                                             // tile_dim1 (tile_dim2=0)
    g1[5] = (int)stride0;                                             // tensor_dim0_stride lo32
    g1[6] = 0;
    g1[7] = 0;
    i32x4 z4 = {0, 0, 0, 0};
    i32x8 z8 = {0, 0, 0, 0, 0, 0, 0, 0};
    __builtin_amdgcn_tensor_load_to_lds(g0, g1, z4, z4, z8, 0);
}

__device__ __forceinline__ unsigned lds_off_of(const void* p) {
    return (unsigned)(uintptr_t)p;
}

// ======================================================================
// bf16 GEMM:  D[MxN] = A[MxK] * Wt[NxK]^T   (Wt is B pre-transposed)
// Block: 256 threads (8 waves), tile M=128, N=64, K-step 32.
// Staging: TDM tensor_load_to_lds, double-buffered, issued by wave 0,
// synchronized with s_wait_tensorcnt + workgroup barriers.
// Each wave: 32x32 sub-tile = 4 v_wmma_f32_16x16x32_bf16 per k-step.
// ======================================================================
template <typename OutT>
__global__ __launch_bounds__(256)
void gemm_bf16wmma(const bf16* __restrict__ A, const bf16* __restrict__ Wt,
                   OutT* __restrict__ D, int M, int N, int K) {
    __shared__ bf16 lsA[2][128][32];   // [buf][m][k]
    __shared__ bf16 lsB[2][64][32];    // [buf][n][k]

    const int tid  = threadIdx.x;
    const int m0   = blockIdx.y * 128;
    const int n0   = blockIdx.x * 64;
    const int ln   = tid & 31;
    const int wv   = tid >> 5;
    const int wm   = (wv >> 1) * 32;
    const int wn   = (wv & 1) * 32;
    const int nidx = ln & 15;
    const int hf   = ln >> 4;
    const int mbase = hf * 8;
    const int kcA  = hf * 8;
    const int kcB  = hf * 8;
    const bool issuer = (wv == 0);

    const bf16* Abase = A  + (size_t)m0 * K;
    const bf16* Bbase = Wt + (size_t)n0 * K;

    v8f acc[2][2] = {};
    const int nk = K >> 5;

    if (issuer) {
        tdm_load_2d(lds_off_of(&lsA[0][0][0]), Abase, K, 128, 32, 128, K);
        tdm_load_2d(lds_off_of(&lsB[0][0][0]), Bbase, K, 64, 32, 64, K);
    }

    for (int i = 0; i < nk; ++i) {
        const int buf = i & 1;
        if (issuer) {
            if (i + 1 < nk) {
                const int kk = (i + 1) << 5;
                tdm_load_2d(lds_off_of(&lsA[buf ^ 1][0][0]), Abase + kk, K, 128, 32, 128, K);
                tdm_load_2d(lds_off_of(&lsB[buf ^ 1][0][0]), Bbase + kk, K, 64, 32, 64, K);
                __builtin_amdgcn_s_wait_tensorcnt(2);  // stage i complete
            } else {
                __builtin_amdgcn_s_wait_tensorcnt(0);
            }
        }
        __syncthreads();   // tile ready for all waves

        BF16x16 af[2], bf[2];
#pragma unroll
        for (int tm = 0; tm < 2; ++tm) {
            int row = wm + tm * 16 + nidx;
            af[tm].h[0] = *(const v8bf*)&lsA[buf][row][kcA];
            af[tm].h[1] = *(const v8bf*)&lsA[buf][row][kcA + 16];
        }
#pragma unroll
        for (int tn = 0; tn < 2; ++tn) {
            int col = wn + tn * 16 + nidx;
            bf[tn].h[0] = *(const v8bf*)&lsB[buf][col][kcB];
            bf[tn].h[1] = *(const v8bf*)&lsB[buf][col][kcB + 16];
        }
#pragma unroll
        for (int tm = 0; tm < 2; ++tm)
#pragma unroll
            for (int tn = 0; tn < 2; ++tn)
                acc[tm][tn] = wmma_bf16(af[tm].v, bf[tn].v, acc[tm][tn]);

        __syncthreads();   // all reads done before TDM overwrites this buffer
    }

    // epilogue (C/D layout: lane&15 = N, rows mbase+r)
#pragma unroll
    for (int tm = 0; tm < 2; ++tm)
#pragma unroll
        for (int tn = 0; tn < 2; ++tn)
#pragma unroll
            for (int r = 0; r < 8; ++r) {
                int row = m0 + wm + tm * 16 + mbase + r;
                int col = n0 + wn + tn * 16 + nidx;
                store_out(&D[(size_t)row * N + col], acc[tm][tn][r]);
            }
}

// ======================================================================
// Prep: f32 -> bf16 (row-major copy)
// ======================================================================
__global__ __launch_bounds__(256)
void cvt_bf16(const float* __restrict__ src, bf16* __restrict__ dst, int n) {
    int i = blockIdx.x * 256 + threadIdx.x;
    if (i < n) dst[i] = f2bf(src[i]);
}

// Prep: f32 [K][N] -> bf16 [N][K] (transpose for TDM-friendly B tiles)
__global__ __launch_bounds__(256)
void transpose_cvt(const float* __restrict__ w, bf16* __restrict__ wt,
                   int K, int N) {
    int i = blockIdx.x * 256 + threadIdx.x;
    if (i >= K * N) return;
    int k = i / N, n = i - k * N;
    wt[(size_t)n * K + k] = f2bf(w[i]);
}

// ======================================================================
// RoPE + rearrange (from bf16 qkv):
//  qs [BH][T][64] (pre-scaled 1/sqrt(64)), ks [BH][T][64], vt [BH][64][T]
// ======================================================================
__global__ __launch_bounds__(256)
void rope_rearrange(const bf16* __restrict__ qkv,
                    bf16* __restrict__ qs, bf16* __restrict__ ks,
                    bf16* __restrict__ vt) {
    int i = blockIdx.x * 256 + threadIdx.x;
    if (i >= MM * CC) return;
    int c   = i & (CC - 1);
    int row = i >> 10;
    int b   = row >> 11;
    int t   = row & (TT - 1);
    int h   = c >> 6;
    int d   = c & 63;
    int bh  = b * HH + h;

    const bf16* base = qkv + (size_t)row * (3 * CC);
    float qv = (float)base[c];
    float kv = (float)base[CC + c];
    float vv = (float)base[2 * CC + c];

    float theta = __powf(10000.0f, -(float)(d & 31) * (1.0f / 32.0f));
    float sn, cs;
    __sincosf((float)t * theta, &sn, &cs);

    int dpair = (d < 32) ? (d + 32) : (d - 32);
    float qp = (float)base[c - d + dpair];
    float kp = (float)base[CC + c - d + dpair];
    float qrot = (d < 32) ? -qp : qp;
    float krot = (d < 32) ? -kp : kp;

    float qo = (qv * cs + qrot * sn) * 0.125f;
    float ko =  kv * cs + krot * sn;

    size_t qkidx = ((size_t)bh * TT + t) * HD + d;
    qs[qkidx] = f2bf(qo);
    ks[qkidx] = f2bf(ko);
    vt[((size_t)bh * HD + d) * TT + t] = f2bf(vv);
}

// ======================================================================
// Flash attention: 1 wave/block, 16 queries/wave, 32-key steps,
// online softmax, P relayout C->A via 1KB LDS, 8 WMMA per step.
// Output bf16 [B][T][C] (feeds final TDM-staged GEMM).
// ======================================================================
__global__ __launch_bounds__(32)
void flash_attn(const bf16* __restrict__ qs, const bf16* __restrict__ ks,
                const bf16* __restrict__ vt, bf16* __restrict__ attn) {
    __shared__ bf16 lsP[16][32];

    const int q0   = blockIdx.x * 16;
    const int bh   = blockIdx.y;
    const int ln   = threadIdx.x & 31;
    const int nidx = ln & 15;
    const int hf   = ln >> 4;
    const int mbase = hf * 8;
    const int kcA  = hf * 8;
    const int kcB  = hf * 8;

    const bf16* qb = qs + (size_t)bh * TT * HD;
    const bf16* kb = ks + (size_t)bh * TT * HD;
    const bf16* vb = vt + (size_t)bh * HD * TT;

    BF16x16 qf[2];
    {
        const bf16* qr = qb + (size_t)(q0 + nidx) * HD;
        qf[0].h[0] = *(const v8bf*)(qr + kcA);
        qf[0].h[1] = *(const v8bf*)(qr + kcA + 16);
        qf[1].h[0] = *(const v8bf*)(qr + 32 + kcA);
        qf[1].h[1] = *(const v8bf*)(qr + 32 + kcA + 16);
    }

    v8f acc[4] = {};
    float mi[8], li[8];
#pragma unroll
    for (int r = 0; r < 8; ++r) { mi[r] = -3.0e38f; li[r] = 0.0f; }

    const int jend = (q0 + 16 + 31) >> 5;
    for (int j = 0; j < jend; ++j) {
        const int key0 = j * 32;

        v8f s[2];
#pragma unroll
        for (int f = 0; f < 2; ++f) {
            const bf16* kr = kb + (size_t)(key0 + f * 16 + nidx) * HD;
            BF16x16 b0, b1;
            b0.h[0] = *(const v8bf*)(kr + kcB);
            b0.h[1] = *(const v8bf*)(kr + kcB + 16);
            b1.h[0] = *(const v8bf*)(kr + 32 + kcB);
            b1.h[1] = *(const v8bf*)(kr + 32 + kcB + 16);
            v8f z = {};
            z = wmma_bf16(qf[0].v, b0.v, z);
            z = wmma_bf16(qf[1].v, b1.v, z);
            s[f] = z;
        }

        float alpha[8];
#pragma unroll
        for (int r = 0; r < 8; ++r) {
            int qrow = q0 + mbase + r;
            int k0   = key0 + nidx;
            float s0 = s[0][r]; if (k0      > qrow) s0 = -3.0e38f;
            float s1 = s[1][r]; if (k0 + 16 > qrow) s1 = -3.0e38f;
            float mx = fmaxf(s0, s1);
            mx = fmaxf(mx, __shfl_xor(mx, 1));
            mx = fmaxf(mx, __shfl_xor(mx, 2));
            mx = fmaxf(mx, __shfl_xor(mx, 4));
            mx = fmaxf(mx, __shfl_xor(mx, 8));
            float mnew = fmaxf(mi[r], mx);
            float a  = __expf(mi[r] - mnew);
            float p0 = __expf(s0 - mnew);
            float p1 = __expf(s1 - mnew);
            float rs = p0 + p1;
            rs += __shfl_xor(rs, 1);
            rs += __shfl_xor(rs, 2);
            rs += __shfl_xor(rs, 4);
            rs += __shfl_xor(rs, 8);
            li[r] = li[r] * a + rs;
            mi[r] = mnew;
            alpha[r] = a;
            lsP[mbase + r][nidx]      = f2bf(p0);
            lsP[mbase + r][16 + nidx] = f2bf(p1);
        }

#pragma unroll
        for (int db = 0; db < 4; ++db)
#pragma unroll
            for (int r = 0; r < 8; ++r)
                acc[db][r] *= alpha[r];

        __syncthreads();

        BF16x16 pf;
        {
            const bf16* pr = &lsP[nidx][0];
            pf.h[0] = *(const v8bf*)(pr + kcA);
            pf.h[1] = *(const v8bf*)(pr + kcA + 16);
        }

#pragma unroll
        for (int db = 0; db < 4; ++db) {
            const bf16* vr = vb + (size_t)(db * 16 + nidx) * TT + key0 + kcB;
            BF16x16 vf;
            vf.h[0] = *(const v8bf*)(vr);
            vf.h[1] = *(const v8bf*)(vr + 16);
            acc[db] = wmma_bf16(pf.v, vf.v, acc[db]);
        }
        __syncthreads();
    }

    int b = bh >> 4, h = bh & 15;
#pragma unroll
    for (int db = 0; db < 4; ++db)
#pragma unroll
        for (int r = 0; r < 8; ++r) {
            int qrow = q0 + mbase + r;
            int col  = h * HD + db * 16 + nidx;
            attn[((size_t)(b * TT + qrow)) * CC + col] = f2bf(acc[db][r] / li[r]);
        }
}

// ======================================================================
// Launcher
// ======================================================================
extern "C" void kernel_launch(void* const* d_in, const int* in_sizes, int n_in,
                              void* d_out, int out_size, void* d_ws, size_t ws_size,
                              hipStream_t stream) {
    const float* x     = (const float*)d_in[0];   // [4,2048,1024]
    const float* w_qkv = (const float*)d_in[1];   // [1024,3072]
    const float* w_out = (const float*)d_in[2];   // [1024,1024]
    float* out = (float*)d_out;                   // [4,2048,1024]

    char* ws = (char*)d_ws;
    size_t off = 0;
    bf16* xb     = (bf16*)(ws + off); off += (size_t)MM * CC * 2;          // 16 MB
    bf16* wqkvT  = (bf16*)(ws + off); off += (size_t)3 * CC * CC * 2;      //  6 MB
    bf16* woutT  = (bf16*)(ws + off); off += (size_t)CC * CC * 2;          //  2 MB
    bf16* qkv    = (bf16*)(ws + off); off += (size_t)MM * 3 * CC * 2;      // 48 MB
    bf16* q_s    = (bf16*)(ws + off); off += (size_t)MM * CC * 2;          // 16 MB
    bf16* k_s    = (bf16*)(ws + off); off += (size_t)MM * CC * 2;          // 16 MB
    bf16* v_t    = (bf16*)(ws + off); off += (size_t)MM * CC * 2;          // 16 MB
    bf16* attnb  = (bf16*)(ws + off);                                      // 16 MB

    // 0) one-time precision/layout prep
    int nx = MM * CC;
    cvt_bf16<<<(nx + 255) / 256, 256, 0, stream>>>(x, xb, nx);
    int nq = CC * 3 * CC;
    transpose_cvt<<<(nq + 255) / 256, 256, 0, stream>>>(w_qkv, wqkvT, CC, 3 * CC);
    int no = CC * CC;
    transpose_cvt<<<(no + 255) / 256, 256, 0, stream>>>(w_out, woutT, CC, CC);

    // 1) QKV projection (TDM-staged bf16 WMMA)
    dim3 g1((3 * CC) / 64, MM / 128);
    gemm_bf16wmma<bf16><<<g1, 256, 0, stream>>>(xb, wqkvT, qkv, MM, 3 * CC, CC);

    // 2) RoPE + rearrange
    rope_rearrange<<<(nx + 255) / 256, 256, 0, stream>>>(qkv, q_s, k_s, v_t);

    // 3) Flash attention
    dim3 g2(TT / 16, BB * HH);
    flash_attn<<<g2, 32, 0, stream>>>(q_s, k_s, v_t, attnb);

    // 4) Output projection -> fp32 result
    dim3 g3(CC / 64, MM / 128);
    gemm_bf16wmma<float><<<g3, 256, 0, stream>>>(attnb, woutT, out, MM, CC, CC);
}
